// LSTM_49374944035648
// MI455X (gfx1250) — compile-verified, matrix operand order checked by reading, not run on hardware
//
#include <hip/hip_runtime.h>
#include <hip/hip_bf16.h>

// LSTM persistent-scan kernel for gfx1250 (MI455X), bf16 WMMA with f32 accum.
//
// Sizes: B=256, T=1024, D=128, H=256, NCLS=10.
// Gate GEMM column order: n = h*4 + gate, gate: 0=g(tanh),1=i,2=f,3=o.
// K order: k in [0,128) -> x part, k in [128,384) -> h part.
//
// Grid: 16 WGs (one per 16-batch-row tile), 512 threads = 16 waves.
// Each wave owns 4 N-tiles (64 columns). Weights are packed once to bf16
// (768 KB, L2-resident) in the exact WMMA B-fragment lane layout and streamed
// per step; an opaque asm zero prevents the compiler from hoisting/spilling.

typedef __bf16 bf16_t;
typedef __attribute__((ext_vector_type(16))) __bf16 v16bf;
typedef __attribute__((ext_vector_type(8)))  __bf16 v8bf;
typedef __attribute__((ext_vector_type(8)))  float  v8f;

#define T_LEN   1024
#define D_IN    128
#define HID     256
#define NCLS    10
#define NKC     12          // 384 / 32 K-chunks
#define TPW     4           // N-tiles per wave
#define NTHR    512         // 16 waves
#define WELEM   (384*1024)  // packed-weight bf16 element count
#define TILE_ELEMS (NKC*512)  // bf16 elems per N-tile group = 6144

union ABu { v16bf v; v8bf h[2]; };

// ---------------------------------------------------------------------------
// Prep kernel: pack [Wx|Wh]^T (K=384 x N=1024) to bf16 in B-fragment layout.
//   fid = ((grp*32 + lane)*16 + j),  grp = ntw*12 + kc,  ntw = global N-tile
//   N = ntw*16 + (lane&15)
//   K = kc*32 + (j>>3)*16 + (lane>>4)*8 + (j&7)
// Each lane's 16 elements are 32 contiguous bytes -> two b128 loads.
// ---------------------------------------------------------------------------
__global__ void pack_weights(const float* __restrict__ Wgx, const float* __restrict__ Wgh,
                             const float* __restrict__ Wix, const float* __restrict__ Wih,
                             const float* __restrict__ Wfx, const float* __restrict__ Wfh,
                             const float* __restrict__ Wox, const float* __restrict__ Woh,
                             bf16_t* __restrict__ out) {
  int fid = blockIdx.x * blockDim.x + threadIdx.x;
  if (fid >= WELEM) return;
  int j    = fid & 15;
  int lane = (fid >> 4) & 31;
  int grp  = fid >> 9;           // 0..767
  int kc   = grp % NKC;
  int ntw  = grp / NKC;          // global N-tile 0..63
  int N    = ntw * 16 + (lane & 15);
  int K    = kc * 32 + ((j >> 3) << 4) + ((lane >> 4) << 3) + (j & 7);
  int h    = N >> 2;
  int g    = N & 3;
  const float* Wx = (g == 0) ? Wgx : (g == 1) ? Wix : (g == 2) ? Wfx : Wox;
  const float* Wh = (g == 0) ? Wgh : (g == 1) ? Wih : (g == 2) ? Wfh : Woh;
  float val = (K < D_IN) ? Wx[h * D_IN + K] : Wh[h * HID + (K - D_IN)];
  out[fid] = (bf16_t)val;
}

// A-matrix LDS swizzle: element (m, k) of the 16x384 bf16 A tile:
//   kc=k/32, r=k%32, p=r/16, half=(r%16)/8, e=r%8
//   idx = kc*512 + p*256 + (half*16 + m)*8 + e      (bf16 units)
__device__ __forceinline__ int aswz_idx(int m, int k) {
  int kc = k >> 5, r = k & 31;
  int p = r >> 4, half = (r & 15) >> 3, e = r & 7;
  return kc * 512 + p * 256 + ((half << 4) + m) * 8 + e;
}

__device__ __forceinline__ float sigm(float xv) {
  return 1.0f / (1.0f + __expf(-xv));
}

__launch_bounds__(NTHR, 4)
__global__ void lstm_persistent(const float* __restrict__ x,
                                const bf16_t* __restrict__ Bw,
                                const float* __restrict__ bg, const float* __restrict__ bi,
                                const float* __restrict__ bf_, const float* __restrict__ bo,
                                const float* __restrict__ Wph, const float* __restrict__ bp,
                                float* __restrict__ out) {
  extern __shared__ char smem[];
  bf16_t* Aswz = (bf16_t*)smem;                 // 12*512 bf16 = 12 KB
  float*  hbuf = (float*)(smem + NKC * 1024);   // 16*256 f32  = 16 KB

  const int tid   = threadIdx.x;
  const int w     = tid >> 5;        // wave 0..15
  const int L     = tid & 31;        // lane
  const int col   = L & 15;
  const int halfL = L >> 4;
  const int b0    = blockIdx.x * 16; // batch-row tile

  // Per-lane column metadata + bias.
  float bias[TPW]; int gate[TPW], hh[TPW];
  #pragma unroll
  for (int nt = 0; nt < TPW; ++nt) {
    int n = w * (TPW * 16) + nt * 16 + col;
    gate[nt] = n & 3;
    hh[nt]   = n >> 2;
    const float* bptr = (gate[nt] == 0) ? bg : (gate[nt] == 1) ? bi
                      : (gate[nt] == 2) ? bf_ : bo;
    bias[nt] = bptr[hh[nt]];
  }

  // init: zero A buffer (h0 = 0), then scatter x_0 into the x slots.
  for (int i = tid; i < NKC * 512; i += NTHR) Aswz[i] = (bf16_t)0.0f;
  {
    int id = tid * 4;                 // 2048 x-elems / 512 threads
    int m = id >> 7, k = id & 127;
    float4 xv = *(const float4*)(x + (size_t)(b0 + m) * T_LEN * D_IN + k);
    Aswz[aswz_idx(m, k + 0)] = (bf16_t)xv.x;
    Aswz[aswz_idx(m, k + 1)] = (bf16_t)xv.y;
    Aswz[aswz_idx(m, k + 2)] = (bf16_t)xv.z;
    Aswz[aswz_idx(m, k + 3)] = (bf16_t)xv.w;
  }
  __syncthreads();

  float c[TPW][8];
  #pragma unroll
  for (int nt = 0; nt < TPW; ++nt)
    #pragma unroll
    for (int j = 0; j < 8; ++j) c[nt][j] = 0.0f;

  const bf16_t* Bwave = Bw + (size_t)w * TPW * TILE_ELEMS;
  const int xm = (tid * 4) >> 7, xk = (tid * 4) & 127;

  for (int t = 0; t < T_LEN; ++t) {
    // Opaque zero: defeats loop-invariant hoisting of the B streams so the
    // compiler re-issues global_load_b128 from the shared packed buffer.
    int z;
    asm volatile("s_mov_b32 %0, 0" : "=s"(z));
    const bf16_t* Bz = Bwave + z;

    // Prefetch x_{t+1} early so its latency overlaps the GEMM.
    float4 xv;
    if (t + 1 < T_LEN)
      xv = *(const float4*)(x + (size_t)(b0 + xm) * T_LEN * D_IN
                              + (size_t)(t + 1) * D_IN + xk);

    v8f acc[TPW];
    #pragma unroll
    for (int nt = 0; nt < TPW; ++nt) acc[nt] = (v8f){0.f,0.f,0.f,0.f,0.f,0.f,0.f,0.f};

    // Software-pipelined K loop: double-buffered A (LDS) + B (global) frags.
    ABu aP[2]; ABu bP[2][TPW];
    aP[0].h[0] = *(const v8bf*)(Aswz + 0 * 512 +       L * 8);
    aP[0].h[1] = *(const v8bf*)(Aswz + 0 * 512 + 256 + L * 8);
    #pragma unroll
    for (int nt = 0; nt < TPW; ++nt) {
      const bf16_t* p = Bz + nt * TILE_ELEMS;
      bP[0][nt].h[0] = *(const v8bf*)(p + L * 16);
      bP[0][nt].h[1] = *(const v8bf*)(p + L * 16 + 8);
    }
    #pragma unroll
    for (int kc = 0; kc < NKC; ++kc) {
      const int cb = kc & 1, nb = cb ^ 1;
      if (kc + 1 < NKC) {
        aP[nb].h[0] = *(const v8bf*)(Aswz + (kc + 1) * 512 +       L * 8);
        aP[nb].h[1] = *(const v8bf*)(Aswz + (kc + 1) * 512 + 256 + L * 8);
        #pragma unroll
        for (int nt = 0; nt < TPW; ++nt) {
          const bf16_t* p = Bz + nt * TILE_ELEMS + (kc + 1) * 512;
          bP[nb][nt].h[0] = *(const v8bf*)(p + L * 16);
          bP[nb][nt].h[1] = *(const v8bf*)(p + L * 16 + 8);
        }
      }
      #pragma unroll
      for (int nt = 0; nt < TPW; ++nt)
        acc[nt] = __builtin_amdgcn_wmma_f32_16x16x32_bf16(
            false, aP[cb].v, false, bP[cb][nt].v, (short)0, acc[nt], false, false);
    }

    // Gate nonlinearities + state update; 4 gates of each hidden unit live in
    // 4 adjacent lanes -> gather with shuffles, keep c in registers.
    float hreg[TPW][8];
    #pragma unroll
    for (int nt = 0; nt < TPW; ++nt) {
      int g = gate[nt];
      #pragma unroll
      for (int j = 0; j < 8; ++j) {
        float v  = acc[nt][j] + bias[nt];
        float v1 = __shfl_xor(v, 1, 32);   // gate g^1
        float v2 = __shfl_xor(v, 2, 32);   // gate g^2
        float v3 = __shfl_xor(v1, 2, 32);  // gate g^3
        float G0 = (g == 0) ? v : (g == 1) ? v1 : (g == 2) ? v2 : v3;
        float G1 = (g == 1) ? v : (g == 0) ? v1 : (g == 3) ? v2 : v3;
        float G2 = (g == 2) ? v : (g == 3) ? v1 : (g == 0) ? v2 : v3;
        float G3 = (g == 3) ? v : (g == 2) ? v1 : (g == 1) ? v2 : v3;
        float gt = tanhf(G0);
        float it = sigm(G1);
        float ft = sigm(G2);
        float ot = sigm(G3);
        float cn = gt * it + c[nt][j] * ft;
        c[nt][j] = cn;
        hreg[nt][j] = tanhf(cn) * ot;
      }
    }

    __syncthreads();  // all A reads of this step are done

    // gate-0 lanes write h back (bf16) into the A buffer's h slots.
    #pragma unroll
    for (int nt = 0; nt < TPW; ++nt) {
      if (gate[nt] == 0) {
        int k = D_IN + hh[nt];
        #pragma unroll
        for (int j = 0; j < 8; ++j) {
          int m = j + halfL * 8;
          Aswz[aswz_idx(m, k)] = (bf16_t)hreg[nt][j];
          if (t == T_LEN - 1) hbuf[m * HID + hh[nt]] = hreg[nt][j];
        }
      }
    }
    // scatter prefetched x_{t+1}
    if (t + 1 < T_LEN) {
      Aswz[aswz_idx(xm, xk + 0)] = (bf16_t)xv.x;
      Aswz[aswz_idx(xm, xk + 1)] = (bf16_t)xv.y;
      Aswz[aswz_idx(xm, xk + 2)] = (bf16_t)xv.z;
      Aswz[aswz_idx(xm, xk + 3)] = (bf16_t)xv.w;
    }
    __syncthreads();
  }

  // final projection: out[b, cls] = h_T @ W_ph^T + b_p  (f32, tiny)
  if (tid < 16 * NCLS) {
    int m = tid / NCLS, cls = tid % NCLS;
    float s = bp[cls];
    #pragma unroll 4
    for (int k = 0; k < HID; ++k) s += hbuf[m * HID + k] * Wph[cls * HID + k];
    out[(size_t)(b0 + m) * NCLS + cls] = s;
  }
}

extern "C" void kernel_launch(void* const* d_in, const int* in_sizes, int n_in,
                              void* d_out, int out_size, void* d_ws, size_t ws_size,
                              hipStream_t stream) {
  const float* x    = (const float*)d_in[0];
  const float* W_gx = (const float*)d_in[1];
  const float* W_gh = (const float*)d_in[2];
  const float* b_g  = (const float*)d_in[3];
  const float* W_ix = (const float*)d_in[4];
  const float* W_ih = (const float*)d_in[5];
  const float* b_i  = (const float*)d_in[6];
  const float* W_fx = (const float*)d_in[7];
  const float* W_fh = (const float*)d_in[8];
  const float* b_f  = (const float*)d_in[9];
  const float* W_ox = (const float*)d_in[10];
  const float* W_oh = (const float*)d_in[11];
  const float* b_o  = (const float*)d_in[12];
  const float* W_ph = (const float*)d_in[13];
  const float* b_p  = (const float*)d_in[14];
  float* outp = (float*)d_out;

  bf16_t* Bw = (bf16_t*)d_ws;  // 768 KB packed bf16 weights

  pack_weights<<<(WELEM + 255) / 256, 256, 0, stream>>>(
      W_gx, W_gh, W_ix, W_ih, W_fx, W_fh, W_ox, W_oh, Bw);

  size_t shmem = NKC * 1024 + 16 * HID * sizeof(float);  // 12 KB + 16 KB
  lstm_persistent<<<16, NTHR, shmem, stream>>>(
      x, Bw, b_g, b_i, b_f, b_o, W_ph, b_p, outp);
}
